// Qwen3MoeSparseMoeBlock_24833500906102
// MI455X (gfx1250) — matrix operand order repaired and checked by reading, not compiled, not run on hardware
//
#include <hip/hip_runtime.h>
#include <hip/hip_bf16.h>

// Problem constants (match reference)
#define T_TOK 2048
#define H_DIM 2048
#define E_EXP 32
#define I_DIM 768
#define K_TOP 8

typedef __attribute__((ext_vector_type(16))) __bf16 v16bf;
typedef __attribute__((ext_vector_type(8)))  __bf16 bf16x8;
typedef __attribute__((ext_vector_type(2)))  __bf16 bf16x2;
typedef __attribute__((ext_vector_type(8)))  float  v8f;
typedef unsigned int u32x4 __attribute__((ext_vector_type(4)));
typedef int          i32x4 __attribute__((ext_vector_type(4)));
typedef int          i32x8 __attribute__((ext_vector_type(8)));

#define TILE_M 128   // rows per block (8 waves x 16)
#define TILE_N 64
#define KT     32
#define LDT    40    // padded LDS row stride in bf16 elements (80B, 16B-aligned rows)

// ---- WMMA B fragment load from LDS (Bs[n][k] rows, stride LDT) ----
__device__ __forceinline__ v16bf frag_b(const __bf16* row, int half) {
  bf16x8 c0 = *(const bf16x8*)(row + half * 8);
  bf16x8 c1 = *(const bf16x8*)(row + 16 + half * 8);
  return __builtin_shufflevector(c0, c1, 0,1,2,3,4,5,6,7,8,9,10,11,12,13,14,15);
}
// ---- WMMA A fragment from 16 f32 staged in registers ----
__device__ __forceinline__ v16bf frag_a_cvt(float4 r0, float4 r1, float4 r2, float4 r3) {
  v16bf a;
  a[0]=(__bf16)r0.x; a[1]=(__bf16)r0.y; a[2]=(__bf16)r0.z; a[3]=(__bf16)r0.w;
  a[4]=(__bf16)r1.x; a[5]=(__bf16)r1.y; a[6]=(__bf16)r1.z; a[7]=(__bf16)r1.w;
  a[8]=(__bf16)r2.x; a[9]=(__bf16)r2.y; a[10]=(__bf16)r2.z; a[11]=(__bf16)r2.w;
  a[12]=(__bf16)r3.x; a[13]=(__bf16)r3.y; a[14]=(__bf16)r3.z; a[15]=(__bf16)r3.w;
  return a;
}

// ---- Tensor Data Mover: async 2-D f32 tile load into LDS ----
__device__ __forceinline__ void tdm_load_tile_f32(unsigned lds_off, const float* gsrc,
                                                  unsigned tile_d0, unsigned tile_d1,
                                                  unsigned long long stride0_elems) {
  unsigned long long ga = (unsigned long long)(uintptr_t)gsrc;
  u32x4 g0;
  g0[0] = 0x1u;                                  // count=1, user descriptor
  g0[1] = lds_off;                               // lds_addr (bytes)
  g0[2] = (unsigned)(ga & 0xFFFFFFFFu);          // global_addr[31:0]
  g0[3] = (unsigned)((ga >> 32) & 0x01FFFFFFu)   // global_addr[56:32]
        | (2u << 30);                            // type=2 ("image")
  i32x8 g1;
  g1[0] = (int)(2u << 16);                       // workgroup_mask=0, data_size=4B
  g1[1] = (int)((tile_d0 & 0xFFFFu) << 16);      // tensor_dim0.lo (tile in-bounds)
  g1[2] = (int)((tile_d0 >> 16) | ((tile_d1 & 0xFFFFu) << 16));
  g1[3] = (int)((tile_d1 >> 16) | (tile_d0 << 16)); // tensor_dim1.hi | tile_dim0
  g1[4] = (int)(tile_d1 & 0xFFFFu);              // tile_dim1 | tile_dim2=0
  g1[5] = (int)(unsigned)(stride0_elems & 0xFFFFFFFFull);     // dim0_stride.lo
  g1[6] = (int)(unsigned)((stride0_elems >> 32) & 0xFFFFull); // dim0_stride.hi
  g1[7] = 0;                                     // dim1_stride=0 (2-D)
  i32x4 z4; z4[0] = 0; z4[1] = 0; z4[2] = 0; z4[3] = 0;
#if defined(__clang_major__) && (__clang_major__ >= 23)
  i32x8 z8; z8[0]=0; z8[1]=0; z8[2]=0; z8[3]=0; z8[4]=0; z8[5]=0; z8[6]=0; z8[7]=0;
  __builtin_amdgcn_tensor_load_to_lds(g0, g1, z4, z4, z8, 0);
#else
  __builtin_amdgcn_tensor_load_to_lds(g0, g1, z4, z4, 0);
#endif
}
__device__ __forceinline__ unsigned lds_offset_of(const void* p) {
  // Generic LDS addresses are {SHARED_BASE, offset[31:0]} per ISA aperture rules.
  return (unsigned)(uintptr_t)p;
}

// ---------------------------------------------------------------------------
// Kernel 1: router. One wave per token: logits -> softmax -> top-8 -> renorm.
// ---------------------------------------------------------------------------
__global__ __launch_bounds__(256)
void moe_router(const float* __restrict__ x, const float* __restrict__ wg,
                int* __restrict__ topk_idx, float* __restrict__ topk_w) {
  const int lane = threadIdx.x & 31;
  const int wid  = threadIdx.x >> 5;
  const int t    = blockIdx.x * 8 + wid;

  float acc = 0.f;
  const float* xr = x + (size_t)t * H_DIM;
  for (int h = 0; h < H_DIM; h += 4) {
    float4 xv = *(const float4*)(xr + h);
    acc += xv.x * wg[(h + 0) * E_EXP + lane];
    acc += xv.y * wg[(h + 1) * E_EXP + lane];
    acc += xv.z * wg[(h + 2) * E_EXP + lane];
    acc += xv.w * wg[(h + 3) * E_EXP + lane];
  }
  float m = acc;
  for (int off = 16; off > 0; off >>= 1) m = fmaxf(m, __shfl_xor(m, off, 32));
  float ex = __expf(acc - m);
  float s = ex;
  for (int off = 16; off > 0; off >>= 1) s += __shfl_xor(s, off, 32);
  float p = ex / s;

  float myw = 0.f; int myi = 0; float wsum = 0.f;
  for (int j = 0; j < K_TOP; ++j) {
    float v = p; int id = lane;
    for (int off = 16; off > 0; off >>= 1) {
      float ov = __shfl_xor(v, off, 32);
      int  oid = __shfl_xor(id, off, 32);
      if (ov > v || (ov == v && oid < id)) { v = ov; id = oid; }
    }
    if (lane == j) { myw = v; myi = id; }
    wsum += v;
    if (lane == id) p = -1.0f;
  }
  if (lane < K_TOP) {
    topk_w[t * K_TOP + lane]   = myw / wsum;
    topk_idx[t * K_TOP + lane] = myi;
  }
}

// ---------------------------------------------------------------------------
// Kernel 2: build per-expert token lists (deterministic, token-ordered).
// ---------------------------------------------------------------------------
__global__ __launch_bounds__(32)
void moe_build(const int* __restrict__ topk_idx, const float* __restrict__ topk_w,
               int* __restrict__ etok, float* __restrict__ ewt, int* __restrict__ ecnt) {
  const int e = blockIdx.x;
  const int lane = threadIdx.x;
  int count = 0;
  for (int c = 0; c < T_TOK; c += 32) {
    const int t = c + lane;
    bool sel = false; float w = 0.f;
#pragma unroll
    for (int j = 0; j < K_TOP; ++j) {
      if (topk_idx[t * K_TOP + j] == e) { sel = true; w = topk_w[t * K_TOP + j]; }
    }
    unsigned mask = (unsigned)__ballot(sel);
    if (sel) {
      int pos = count + __popc(mask & ((1u << lane) - 1u));
      etok[e * T_TOK + pos] = t;
      ewt [e * T_TOK + pos] = w;
    }
    count += __popc(mask);
  }
  if (lane == 0) ecnt[e] = count;
}

// ---------------------------------------------------------------------------
// Kernel 3: exclusive scan of expert counts (one wave).
// ---------------------------------------------------------------------------
__global__ __launch_bounds__(32)
void moe_scan(const int* __restrict__ ecnt, int* __restrict__ eoff) {
  const int lane = threadIdx.x;
  int c = ecnt[lane];
  int s = c;
  for (int off = 1; off < 32; off <<= 1) {
    int n = __shfl_up(s, off, 32);
    if (lane >= off) s += n;
  }
  eoff[lane] = s - c;
}

// ---------------------------------------------------------------------------
// Kernel 4: zero the output.
// ---------------------------------------------------------------------------
__global__ __launch_bounds__(256)
void moe_zero(float* __restrict__ out) {
  size_t i = ((size_t)blockIdx.x * 256 + threadIdx.x) * 4;
  float4 z = {0.f, 0.f, 0.f, 0.f};
  *(float4*)(out + i) = z;
}

// ---------------------------------------------------------------------------
// Kernel 5: fused gate+up grouped GEMM + SiLU.
// Tile 128x64, k-step 32; 8 waves. TDM double-buffered f32 weight staging with
// bf16x2-packed transposed repack; A rows loaded per-lane direct from global
// (no LDS round trip), software-pipelined one k-step ahead.
// grid = (I/64, T/128, E)
// ---------------------------------------------------------------------------
__global__ __launch_bounds__(256)
void moe_gateup(const float* __restrict__ x, const float* __restrict__ wgp,
                const float* __restrict__ wup, const int* __restrict__ etok,
                const int* __restrict__ ecnt, const int* __restrict__ eoff,
                float* __restrict__ act) {
  const int e  = blockIdx.z;
  const int m0 = blockIdx.y * TILE_M;
  const int n0 = blockIdx.x * TILE_N;
  const int cnt = ecnt[e];
  if (m0 >= cnt) return;
  const int rows = min(TILE_M, cnt - m0);
  const int oe = eoff[e];

  __shared__ __align__(16) __bf16 Bg[TILE_N * LDT];
  __shared__ __align__(16) __bf16 Bu[TILE_N * LDT];
  __shared__ __align__(16) float  StgG[2][KT * TILE_N];  // TDM staging (f32)
  __shared__ __align__(16) float  StgU[2][KT * TILE_N];
  __shared__ int Ts[TILE_M];

  const int tid  = threadIdx.x;
  const int lane = tid & 31;
  const int wid  = tid >> 5;
  const int half = lane >> 4;
  const int l15  = lane & 15;

  const float* wgb = wgp + (size_t)e * H_DIM * I_DIM + n0;
  const float* wub = wup + (size_t)e * H_DIM * I_DIM + n0;

  if (tid < TILE_M) Ts[tid] = etok[e * T_TOK + m0 + min(tid, rows - 1)];
  __syncthreads();

  // Per-lane A row base (this lane's token row, at its half's k-chunk).
  // Clamped rows produce garbage discarded by the m<rows epilogue guard.
  const float* arow =
      x + (size_t)Ts[min(wid * 16 + l15, rows - 1)] * H_DIM + half * 8;

  // Repack unit for this thread: k-row pair kr2, 4 n columns at nc.
  const int kr2 = tid >> 4;        // 0..15 (pairs of k rows)
  const int nc  = (tid & 15) * 4;  // 0..60

  v8f accg[4] = {};
  v8f accu[4] = {};
  const int NK = H_DIM / KT;

  if (wid == 0) {
    tdm_load_tile_f32(lds_offset_of(&StgG[0][0]), wgb, TILE_N, KT, I_DIM);
    tdm_load_tile_f32(lds_offset_of(&StgU[0][0]), wub, TILE_N, KT, I_DIM);
  }
  float4 rA0 = *(const float4*)(arow + 0);
  float4 rA1 = *(const float4*)(arow + 4);
  float4 rA2 = *(const float4*)(arow + 16);
  float4 rA3 = *(const float4*)(arow + 20);

  for (int kt = 0; kt < NK; ++kt) {
    const int cur = kt & 1, nxt = cur ^ 1;
    if (wid == 0) __builtin_amdgcn_s_wait_tensorcnt(0);  // stage[cur] landed
    __syncthreads();  // publish stage[cur]; Bg/Bu free (prev compute done)
    if (wid == 0 && kt + 1 < NK) {  // next DMA overlaps repack+compute
      tdm_load_tile_f32(lds_offset_of(&StgG[nxt][0]),
                        wgb + (size_t)(kt + 1) * KT * I_DIM, TILE_N, KT, I_DIM);
      tdm_load_tile_f32(lds_offset_of(&StgU[nxt][0]),
                        wub + (size_t)(kt + 1) * KT * I_DIM, TILE_N, KT, I_DIM);
    }
    // A fragment for this k-step from staged registers
    v16bf af = frag_a_cvt(rA0, rA1, rA2, rA3);
    // Stage next A chunk (overlaps repack + WMMA)
    if (kt + 1 < NK) {
      const float* an = arow + (size_t)(kt + 1) * KT;
      rA0 = *(const float4*)(an + 0);
      rA1 = *(const float4*)(an + 4);
      rA2 = *(const float4*)(an + 16);
      rA3 = *(const float4*)(an + 20);
    }
    // B repack: f32 staging -> bf16 transposed tiles, bf16x2-packed stores
    {
      float4 g0 = *(const float4*)(&StgG[cur][(2 * kr2 + 0) * TILE_N + nc]);
      float4 g1 = *(const float4*)(&StgG[cur][(2 * kr2 + 1) * TILE_N + nc]);
      float4 u0 = *(const float4*)(&StgU[cur][(2 * kr2 + 0) * TILE_N + nc]);
      float4 u1 = *(const float4*)(&StgU[cur][(2 * kr2 + 1) * TILE_N + nc]);
      bf16x2 p;
      p[0]=(__bf16)g0.x; p[1]=(__bf16)g1.x; *(bf16x2*)(&Bg[(nc+0)*LDT + 2*kr2]) = p;
      p[0]=(__bf16)g0.y; p[1]=(__bf16)g1.y; *(bf16x2*)(&Bg[(nc+1)*LDT + 2*kr2]) = p;
      p[0]=(__bf16)g0.z; p[1]=(__bf16)g1.z; *(bf16x2*)(&Bg[(nc+2)*LDT + 2*kr2]) = p;
      p[0]=(__bf16)g0.w; p[1]=(__bf16)g1.w; *(bf16x2*)(&Bg[(nc+3)*LDT + 2*kr2]) = p;
      p[0]=(__bf16)u0.x; p[1]=(__bf16)u1.x; *(bf16x2*)(&Bu[(nc+0)*LDT + 2*kr2]) = p;
      p[0]=(__bf16)u0.y; p[1]=(__bf16)u1.y; *(bf16x2*)(&Bu[(nc+1)*LDT + 2*kr2]) = p;
      p[0]=(__bf16)u0.z; p[1]=(__bf16)u1.z; *(bf16x2*)(&Bu[(nc+2)*LDT + 2*kr2]) = p;
      p[0]=(__bf16)u0.w; p[1]=(__bf16)u1.w; *(bf16x2*)(&Bu[(nc+3)*LDT + 2*kr2]) = p;
    }
    __syncthreads();

    // Batch all B fragment loads, then the WMMA chain (lets compiler batch waits)
    v16bf bgf[4], buf[4];
#pragma unroll
    for (int nf = 0; nf < 4; ++nf) bgf[nf] = frag_b(&Bg[(nf * 16 + l15) * LDT], half);
#pragma unroll
    for (int nf = 0; nf < 4; ++nf) buf[nf] = frag_b(&Bu[(nf * 16 + l15) * LDT], half);
#pragma unroll
    for (int nf = 0; nf < 4; ++nf) {
      accg[nf] = __builtin_amdgcn_wmma_f32_16x16x32_bf16(
          false, af, false, bgf[nf], (short)0, accg[nf], false, false);
      accu[nf] = __builtin_amdgcn_wmma_f32_16x16x32_bf16(
          false, af, false, buf[nf], (short)0, accu[nf], false, false);
    }
  }

  // Epilogue: SiLU(gate)*up -> compact activation buffer
#pragma unroll
  for (int nf = 0; nf < 4; ++nf) {
#pragma unroll
    for (int vr = 0; vr < 8; ++vr) {
      int m = wid * 16 + vr + half * 8;
      if (m < rows) {
        float g = accg[nf][vr];
        float u = accu[nf][vr];
        float a = (g / (1.f + __expf(-g))) * u;
        act[(size_t)(oe + m0 + m) * I_DIM + n0 + nf * 16 + l15] = a;
      }
    }
  }
}

// ---------------------------------------------------------------------------
// Kernel 6: down-proj grouped GEMM + weighted combine into out (fp32 atomics).
// grid = (H/64, T/128, E)
// ---------------------------------------------------------------------------
__global__ __launch_bounds__(256)
void moe_down(const float* __restrict__ act, const float* __restrict__ wdp,
              const int* __restrict__ etok, const float* __restrict__ ewt,
              const int* __restrict__ ecnt, const int* __restrict__ eoff,
              float* __restrict__ out) {
  const int e  = blockIdx.z;
  const int m0 = blockIdx.y * TILE_M;
  const int n0 = blockIdx.x * TILE_N;
  const int cnt = ecnt[e];
  if (m0 >= cnt) return;
  const int rows = min(TILE_M, cnt - m0);
  const int oe = eoff[e];

  __shared__ __align__(16) __bf16 Bs[TILE_N * LDT];
  __shared__ __align__(16) float  Stg[2][KT * TILE_N];
  __shared__ int   Ts[TILE_M];
  __shared__ float Ws[TILE_M];

  const int tid  = threadIdx.x;
  const int lane = tid & 31;
  const int wid  = tid >> 5;
  const int half = lane >> 4;
  const int l15  = lane & 15;

  if (tid < TILE_M) {
    int rc = min(tid, rows - 1);
    Ts[tid] = etok[e * T_TOK + m0 + rc];
    Ws[tid] = (tid < rows) ? ewt[e * T_TOK + m0 + tid] : 0.f;
  }
  __syncthreads();

  const float* wdb = wdp + (size_t)e * I_DIM * H_DIM + n0;
  // Per-lane A row base into the compact activation buffer (clamped).
  const float* arow =
      act + (size_t)(oe + m0 + min(wid * 16 + l15, rows - 1)) * I_DIM + half * 8;

  const int kr2 = tid >> 4;
  const int nc  = (tid & 15) * 4;

  v8f acc[4] = {};
  const int NK = I_DIM / KT;

  if (wid == 0)
    tdm_load_tile_f32(lds_offset_of(&Stg[0][0]), wdb, TILE_N, KT, H_DIM);
  float4 rA0 = *(const float4*)(arow + 0);
  float4 rA1 = *(const float4*)(arow + 4);
  float4 rA2 = *(const float4*)(arow + 16);
  float4 rA3 = *(const float4*)(arow + 20);

  for (int kt = 0; kt < NK; ++kt) {
    const int cur = kt & 1, nxt = cur ^ 1;
    if (wid == 0) __builtin_amdgcn_s_wait_tensorcnt(0);
    __syncthreads();
    if (wid == 0 && kt + 1 < NK)
      tdm_load_tile_f32(lds_offset_of(&Stg[nxt][0]),
                        wdb + (size_t)(kt + 1) * KT * H_DIM, TILE_N, KT, H_DIM);
    v16bf af = frag_a_cvt(rA0, rA1, rA2, rA3);
    if (kt + 1 < NK) {
      const float* an = arow + (size_t)(kt + 1) * KT;
      rA0 = *(const float4*)(an + 0);
      rA1 = *(const float4*)(an + 4);
      rA2 = *(const float4*)(an + 16);
      rA3 = *(const float4*)(an + 20);
    }
    {
      float4 b0 = *(const float4*)(&Stg[cur][(2 * kr2 + 0) * TILE_N + nc]);
      float4 b1 = *(const float4*)(&Stg[cur][(2 * kr2 + 1) * TILE_N + nc]);
      bf16x2 p;
      p[0]=(__bf16)b0.x; p[1]=(__bf16)b1.x; *(bf16x2*)(&Bs[(nc+0)*LDT + 2*kr2]) = p;
      p[0]=(__bf16)b0.y; p[1]=(__bf16)b1.y; *(bf16x2*)(&Bs[(nc+1)*LDT + 2*kr2]) = p;
      p[0]=(__bf16)b0.z; p[1]=(__bf16)b1.z; *(bf16x2*)(&Bs[(nc+2)*LDT + 2*kr2]) = p;
      p[0]=(__bf16)b0.w; p[1]=(__bf16)b1.w; *(bf16x2*)(&Bs[(nc+3)*LDT + 2*kr2]) = p;
    }
    __syncthreads();

    v16bf bf[4];
#pragma unroll
    for (int nf = 0; nf < 4; ++nf) bf[nf] = frag_b(&Bs[(nf * 16 + l15) * LDT], half);
#pragma unroll
    for (int nf = 0; nf < 4; ++nf)
      acc[nf] = __builtin_amdgcn_wmma_f32_16x16x32_bf16(
          false, af, false, bf[nf], (short)0, acc[nf], false, false);
  }

  // Epilogue: scale by routing weight, combine via fp32 global atomics.
#pragma unroll
  for (int nf = 0; nf < 4; ++nf) {
#pragma unroll
    for (int vr = 0; vr < 8; ++vr) {
      int m = wid * 16 + vr + half * 8;
      if (m < rows) {
        float v = acc[nf][vr] * Ws[m];
        int t = Ts[m];
        float* dst = out + (size_t)t * H_DIM + n0 + nf * 16 + l15;
        __hip_atomic_fetch_add(dst, v, __ATOMIC_RELAXED, __HIP_MEMORY_SCOPE_AGENT);
      }
    }
  }
}

// ---------------------------------------------------------------------------
extern "C" void kernel_launch(void* const* d_in, const int* in_sizes, int n_in,
                              void* d_out, int out_size, void* d_ws, size_t ws_size,
                              hipStream_t stream) {
  (void)in_sizes; (void)n_in; (void)out_size; (void)ws_size;
  const float* x   = (const float*)d_in[0];  // [T,H]
  const float* wg  = (const float*)d_in[1];  // [H,E]
  const float* wgp = (const float*)d_in[2];  // [E,H,I]
  const float* wup = (const float*)d_in[3];  // [E,H,I]
  const float* wdp = (const float*)d_in[4];  // [E,I,H]
  float* out = (float*)d_out;                // [T,H]

  char* p = (char*)d_ws;
  size_t o = 0;
  auto carve = [&](size_t bytes) -> char* {
    char* r = p + o;
    o = (o + bytes + 255) & ~(size_t)255;
    return r;
  };
  int*   topk_idx = (int*)  carve((size_t)T_TOK * K_TOP * 4);
  float* topk_w   = (float*)carve((size_t)T_TOK * K_TOP * 4);
  int*   etok     = (int*)  carve((size_t)E_EXP * T_TOK * 4);
  float* ewt      = (float*)carve((size_t)E_EXP * T_TOK * 4);
  int*   ecnt     = (int*)  carve((size_t)E_EXP * 4);
  int*   eoff     = (int*)  carve((size_t)E_EXP * 4);
  float* act      = (float*)carve((size_t)T_TOK * K_TOP * I_DIM * 4);  // ~50MB

  moe_router<<<dim3(T_TOK / 8), 256, 0, stream>>>(x, wg, topk_idx, topk_w);
  moe_build <<<dim3(E_EXP), 32, 0, stream>>>(topk_idx, topk_w, etok, ewt, ecnt);
  moe_scan  <<<dim3(1), 32, 0, stream>>>(ecnt, eoff);
  moe_zero  <<<dim3((T_TOK * H_DIM) / 1024), 256, 0, stream>>>(out);
  moe_gateup<<<dim3(I_DIM / TILE_N, T_TOK / TILE_M, E_EXP), 256, 0, stream>>>(
      x, wgp, wup, etok, ecnt, eoff, act);
  moe_down  <<<dim3(H_DIM / TILE_N, T_TOK / TILE_M, E_EXP), 256, 0, stream>>>(
      act, wdp, etok, ewt, ecnt, eoff, out);
}
